// OccurrencePool_61632780697735
// MI455X (gfx1250) — compile-verified, matrix operand order checked by reading, not run on hardware
//
#include <hip/hip_runtime.h>
#include <hip/hip_bf16.h>

// ---------------- types ----------------
typedef __bf16 bf16_t;
typedef __attribute__((ext_vector_type(2)))  bf16_t v2bf;
typedef __attribute__((ext_vector_type(8)))  bf16_t v8bf;
typedef __attribute__((ext_vector_type(16))) bf16_t v16bf;
typedef __attribute__((ext_vector_type(8)))  float  v8f;
typedef __attribute__((ext_vector_type(4)))  float  f32x4;
typedef __attribute__((ext_vector_type(4)))  unsigned u32x4;
typedef __attribute__((ext_vector_type(4)))  unsigned short u16x4;
typedef __attribute__((ext_vector_type(8)))  unsigned short u16x8;

// ---------------- helpers ----------------
__device__ __forceinline__ unsigned short f2bf(float f) {
    bf16_t h = (bf16_t)f;                       // RTNE convert (hw cvt on gfx1250)
    return __builtin_bit_cast(unsigned short, h);
}
__device__ __forceinline__ unsigned pack2f(float a, float b) {   // (k,k+1): a -> [15:0]
    v2bf v = { (bf16_t)a, (bf16_t)b };
    return __builtin_bit_cast(unsigned, v);
}
__device__ __forceinline__ unsigned pack2u(unsigned short a, unsigned short b) {
    return (unsigned)a | ((unsigned)b << 16);
}

// Fragment from LDS row. 16-bit A/B layout: lane<16 -> K k0..k0+7 (v0-3) and
// K 16+k0..16+k0+7 (v4-7); k0 = 8*(lane>=16). Pairs (k,k+1) packed per VGPR.
__device__ __forceinline__ v16bf frag_from_lds(const unsigned short* row, int k0) {
    const v8bf lo = *reinterpret_cast<const v8bf*>(row + k0);
    const v8bf hi = *reinterpret_cast<const v8bf*>(row + k0 + 16);
    return __builtin_shufflevector(lo, hi, 0,1,2,3,4,5,6,7,8,9,10,11,12,13,14,15);
}

constexpr int BN = 128, BK = 32, PITCH = 40; // pitch 40 ushorts = 80B, 16B aligned, breaks bank conflicts

// ---------------- generic GEMM: Out(MxN,bf16) = epi(W(MxK,f32) * B(KxN,TB) + bias) ----------------
// EPI: 0 = bias, 1 = bias+relu, 2 = abs (no bias).  Tile: (64*WM) x 128, 8 waves (4M x 2N).
template <typename TB, int EPI, int WM>
__global__ __launch_bounds__(256) void gemm_wmma(
    const float* __restrict__ A, const TB* __restrict__ Bm,
    const float* __restrict__ bias, unsigned short* __restrict__ Out,
    int M, int N, int K, size_t strideB, size_t strideOut)
{
    constexpr int BMt = 64 * WM;
    __shared__ __align__(16) unsigned short lA[BMt * PITCH];
    __shared__ __align__(16) unsigned short lB[BN * PITCH];

    const int tid  = threadIdx.x;
    const int wid  = tid >> 5, lane = tid & 31;
    const int half = lane >> 4, l15 = lane & 15;
    const int waveM = wid & 3, waveN = wid >> 2;
    const int m0 = blockIdx.y * BMt, n0 = blockIdx.x * BN;
    const TB* Bp = Bm + (size_t)blockIdx.z * strideB;
    unsigned short* Op = Out + (size_t)blockIdx.z * strideOut;

    // register staging (software pipeline: next tile's loads overlap WMMA)
    f32x4 aR[WM][2];
    f32x4 bRF[2][2];
    u16x4 bRU[2][2];

    auto gload = [&](int kt) {
        #pragma unroll
        for (int i = 0; i < WM; ++i) {              // A: chunks of 8 along k
            int c = tid + i * 256, m = c >> 2, kc = (c & 3) * 8;
            const float* s = &A[(size_t)(m0 + m) * K + kt + kc];
            aR[i][0] = *reinterpret_cast<const f32x4*>(s);
            aR[i][1] = *reinterpret_cast<const f32x4*>(s + 4);
        }
        #pragma unroll
        for (int i = 0; i < 2; ++i) {               // B: chunks of 4n x 2k
            int c = tid + i * 256, nc = (c & 31) * 4, k = (c >> 5) * 2;
            if constexpr (sizeof(TB) == 4) {
                const float* s = (const float*)&Bp[(size_t)(kt + k) * N + n0 + nc];
                bRF[i][0] = *reinterpret_cast<const f32x4*>(s);
                bRF[i][1] = *reinterpret_cast<const f32x4*>(s + N);
            } else {
                const unsigned short* s = (const unsigned short*)&Bp[(size_t)(kt + k) * N + n0 + nc];
                bRU[i][0] = *reinterpret_cast<const u16x4*>(s);
                bRU[i][1] = *reinterpret_cast<const u16x4*>(s + N);
            }
        }
    };
    auto lstore = [&]() {
        #pragma unroll
        for (int i = 0; i < WM; ++i) {              // A row-major [m][k], b128 stores
            int c = tid + i * 256, m = c >> 2, kc = (c & 3) * 8;
            u32x4 p = { pack2f(aR[i][0].x, aR[i][0].y), pack2f(aR[i][0].z, aR[i][0].w),
                        pack2f(aR[i][1].x, aR[i][1].y), pack2f(aR[i][1].z, aR[i][1].w) };
            *reinterpret_cast<u32x4*>(&lA[m * PITCH + kc]) = p;
        }
        #pragma unroll
        for (int i = 0; i < 2; ++i) {               // B transposed to [n][k], (k,k+1) packed b32
            int c = tid + i * 256, nc = (c & 31) * 4, k = (c >> 5) * 2;
            unsigned q[4];
            if constexpr (sizeof(TB) == 4) {
                q[0] = pack2f(bRF[i][0].x, bRF[i][1].x); q[1] = pack2f(bRF[i][0].y, bRF[i][1].y);
                q[2] = pack2f(bRF[i][0].z, bRF[i][1].z); q[3] = pack2f(bRF[i][0].w, bRF[i][1].w);
            } else {
                q[0] = pack2u(bRU[i][0].x, bRU[i][1].x); q[1] = pack2u(bRU[i][0].y, bRU[i][1].y);
                q[2] = pack2u(bRU[i][0].z, bRU[i][1].z); q[3] = pack2u(bRU[i][0].w, bRU[i][1].w);
            }
            #pragma unroll
            for (int nj = 0; nj < 4; ++nj)
                *reinterpret_cast<unsigned*>(&lB[(nc + nj) * PITCH + k]) = q[nj];
        }
    };

    v8f acc[WM][4] = {};
    gload(0);
    for (int kt = 0; kt < K; kt += BK) {
        __syncthreads();
        lstore();
        __syncthreads();
        if (kt + BK < K) gload(kt + BK);            // overlap next tile's loads with WMMA
        const int k0 = half * 8;
        v16bf af[WM];
        #pragma unroll
        for (int wm = 0; wm < WM; ++wm)
            af[wm] = frag_from_lds(&lA[(waveM * 16 * WM + wm * 16 + l15) * PITCH], k0);
        #pragma unroll
        for (int j = 0; j < 4; ++j) {
            v16bf bfr = frag_from_lds(&lB[(waveN * 64 + j * 16 + l15) * PITCH], k0);
            #pragma unroll
            for (int wm = 0; wm < WM; ++wm)
                acc[wm][j] = __builtin_amdgcn_wmma_f32_16x16x32_bf16(
                    false, af[wm], false, bfr, (short)0, acc[wm][j], false, false);
        }
    }

    #pragma unroll
    for (int wm = 0; wm < WM; ++wm) {
        const int mb = m0 + waveM * 16 * WM + wm * 16 + half * 8;
        float bv[8];
        #pragma unroll
        for (int r = 0; r < 8; ++r) bv[r] = (EPI == 2) ? 0.f : bias[mb + r];
        #pragma unroll
        for (int j = 0; j < 4; ++j) {
            int ng = n0 + waveN * 64 + j * 16 + l15;
            #pragma unroll
            for (int r = 0; r < 8; ++r) {           // C/D layout: M = r + 8*half, N = lane&15
                float v = acc[wm][j][r] + bv[r];
                if (EPI == 1) v = v > 0.f ? v : 0.f;
                if (EPI == 2) v = fabsf(v);
                Op[(size_t)(mb + r) * N + ng] = f2bf(v);
            }
        }
    }
}

// ---------------- split-K contraction: partial[z] = occ_b(64xS) * feat_b(512xS)^T over a K chunk --------
__global__ __launch_bounds__(256) void occ_feat_gemm(
    const unsigned short* __restrict__ Occ, const unsigned short* __restrict__ Feat,
    float* __restrict__ Partial, int S, int KS)
{
    __shared__ __align__(16) unsigned short lA[64 * PITCH];
    __shared__ __align__(16) unsigned short lB[128 * PITCH];
    const int tid  = threadIdx.x;
    const int wid  = tid >> 5, lane = tid & 31;
    const int half = lane >> 4, l15 = lane & 15;
    const int waveM = wid & 3, waveN = wid >> 2;
    const int n0 = blockIdx.x * 128;
    const int b  = blockIdx.z / KS, ks = blockIdx.z % KS;
    const int KC = S / KS;
    const unsigned short* Ap = Occ  + (size_t)b * 64  * S + (size_t)ks * KC;
    const unsigned short* Bp = Feat + (size_t)b * 512 * S + (size_t)ks * KC;

    u16x8 aR, bR[2];
    auto gload = [&](int kt) {
        { int c = tid,           m = c >> 2, kc = (c & 3) * 8;      // A: 64x32, k-contiguous
          aR = *reinterpret_cast<const u16x8*>(&Ap[(size_t)m * S + kt + kc]); }
        #pragma unroll
        for (int i = 0; i < 2; ++i) {                               // B: 128x32, k-contiguous
            int c = tid + i * 256, n = c >> 2, kc = (c & 3) * 8;
            bR[i] = *reinterpret_cast<const u16x8*>(&Bp[(size_t)(n0 + n) * S + kt + kc]);
        }
    };
    auto lstore = [&]() {
        { int c = tid, m = c >> 2, kc = (c & 3) * 8;
          *reinterpret_cast<u16x8*>(&lA[m * PITCH + kc]) = aR; }
        #pragma unroll
        for (int i = 0; i < 2; ++i) {
            int c = tid + i * 256, n = c >> 2, kc = (c & 3) * 8;
            *reinterpret_cast<u16x8*>(&lB[n * PITCH + kc]) = bR[i];
        }
    };

    v8f acc[4] = {};
    gload(0);
    for (int kt = 0; kt < KC; kt += 32) {
        __syncthreads();
        lstore();
        __syncthreads();
        if (kt + 32 < KC) gload(kt + 32);
        const int k0 = half * 8;
        v16bf af = frag_from_lds(&lA[(waveM * 16 + l15) * PITCH], k0);
        #pragma unroll
        for (int j = 0; j < 4; ++j) {
            v16bf bfr = frag_from_lds(&lB[(waveN * 64 + j * 16 + l15) * PITCH], k0);
            acc[j] = __builtin_amdgcn_wmma_f32_16x16x32_bf16(
                false, af, false, bfr, (short)0, acc[j], false, false);
        }
    }

    float* Pp = Partial + (size_t)blockIdx.z * (64 * 512);
    #pragma unroll
    for (int j = 0; j < 4; ++j) {
        int ng = n0 + waveN * 64 + j * 16 + l15;
        #pragma unroll
        for (int r = 0; r < 8; ++r) {
            int mg = waveM * 16 + half * 8 + r;
            Pp[(size_t)mg * 512 + ng] = acc[j][r];
        }
    }
}

__global__ __launch_bounds__(256) void reduce_partials(
    const float* __restrict__ P, float* __restrict__ Out, int KS)
{
    int t = blockIdx.x * blockDim.x + threadIdx.x;        // 0 .. 4*64*512-1
    int b = t >> 15, i = t & 32767;                       // 64*512 = 32768
    float s = 0.f;
    for (int ks = 0; ks < KS; ++ks)
        s += P[((size_t)(b * KS + ks) << 15) + i];
    Out[t] = s * (1.0f / 16384.0f);
}

// ---------------- host ----------------
extern "C" void kernel_launch(void* const* d_in, const int* in_sizes, int n_in,
                              void* d_out, int out_size, void* d_ws, size_t ws_size,
                              hipStream_t stream) {
    constexpr int Bn = 4, CIN = 512, C4 = 128, C8 = 64, S = 16384, KS = 16;
    const float* x      = (const float*)d_in[0];
    const float* w_add1 = (const float*)d_in[1];
    const float* b_add1 = (const float*)d_in[2];
    const float* w_add2 = (const float*)d_in[3];
    const float* b_add2 = (const float*)d_in[4];
    const float* w_occ1 = (const float*)d_in[5];
    const float* b_occ1 = (const float*)d_in[6];
    const float* w_occ2 = (const float*)d_in[7];
    const float* b_occ2 = (const float*)d_in[8];
    const float* w_occ3 = (const float*)d_in[9];
    (void)in_sizes; (void)n_in; (void)out_size; (void)ws_size;

    // workspace layout (needs 128 MiB): [0,64Mi)=Y1 then reused, [64Mi,128Mi)=feat
    char* ws = (char*)d_ws;
    unsigned short* Y1   = (unsigned short*)(ws);                      // 4*512*16384 bf16 = 64 MiB
    unsigned short* feat = (unsigned short*)(ws + ((size_t)64 << 20)); // 64 MiB
    unsigned short* H1   = (unsigned short*)(ws);                      // 16 MiB (Y1 dead)
    unsigned short* H2   = (unsigned short*)(ws + ((size_t)16 << 20)); // 8 MiB
    unsigned short* occ  = (unsigned short*)(ws + ((size_t)24 << 20)); // 8 MiB
    float*          part = (float*)        (ws + ((size_t)32 << 20)); // 4*16*64*512 f32 = 8 MiB

    dim3 blk(256);
    // feat path (128x128 tiles)
    gemm_wmma<float, 1, 2><<<dim3(S / BN, CIN / 128, Bn), blk, 0, stream>>>(
        w_add1, x, b_add1, Y1, CIN, S, CIN, (size_t)CIN * S, (size_t)CIN * S);
    gemm_wmma<unsigned short, 0, 2><<<dim3(S / BN, CIN / 128, Bn), blk, 0, stream>>>(
        w_add2, Y1, b_add2, feat, CIN, S, CIN, (size_t)CIN * S, (size_t)CIN * S);
    // occ path
    gemm_wmma<float, 1, 2><<<dim3(S / BN, 1, Bn), blk, 0, stream>>>(
        w_occ1, x, b_occ1, H1, C4, S, CIN, (size_t)CIN * S, (size_t)C4 * S);
    gemm_wmma<unsigned short, 1, 1><<<dim3(S / BN, 1, Bn), blk, 0, stream>>>(
        w_occ2, H1, b_occ2, H2, C8, S, C4, (size_t)C4 * S, (size_t)C8 * S);
    gemm_wmma<unsigned short, 2, 1><<<dim3(S / BN, 1, Bn), blk, 0, stream>>>(
        w_occ3, H2, nullptr, occ, C8, S, C8, (size_t)C8 * S, (size_t)C8 * S);
    // final contraction (split-K, deterministic reduce)
    occ_feat_gemm<<<dim3(CIN / 128, 1, Bn * KS), blk, 0, stream>>>(occ, feat, part, S, KS);
    reduce_partials<<<dim3((Bn * C8 * CIN) / 256), blk, 0, stream>>>(part, (float*)d_out, KS);
}
